// GraphEncoder_20323785244759
// MI455X (gfx1250) — compile-verified, matrix operand order checked by reading, not run on hardware
//
#include <hip/hip_runtime.h>
#include <hip/hip_bf16.h>
#include <math.h>

// ---------------------------------------------------------------------------
// GAT + LSTM encoder for MI455X (gfx1250), wave32 + WMMA bf16.
//   Stage 1: pre(3->32, tanh) + GATConv over 24-joint skeleton  -> X bf16
//   Stage 2: weight/bias prep (f32 -> bf16, b_ih+b_hh)
//   Stage 3: persistent LSTM kernel, 256 steps, v_wmma_f32_16x16x32_bf16,
//            LDS-resident weight slices, double-buffered h, atomic barrier.
//   Tiling: each wave owns 2 M-tiles x 4 gate-tiles -> 8 WMMA per 4 B-frags,
//   so ds_load_b128 : v_wmma ratio is 1:1 (LDS bandwidth balanced vs XDL).
// ---------------------------------------------------------------------------

typedef __bf16 bf16_t;
typedef __attribute__((ext_vector_type(16))) __bf16 v16bf;
typedef __attribute__((ext_vector_type(8)))  __bf16 v8bf;
typedef __attribute__((ext_vector_type(8)))  float  v8f;

#define NB    128                 // batch N
#define LSEQ  256                 // sequence length L
#define NJ    24                  // SMPL joints
#define HIDD  32                  // per-joint hidden
#define HL    768                 // LSTM hidden = NJ*HIDD
#define G4    3072                // 4*HL gate rows
#define NGRAPH (NB*LSEQ)          // 32768 graphs
#define CWG   48                  // LSTM column workgroups (768/16)
#define LDP   (HL+8)              // padded LDS row stride (bf16 elems): 16B aligned,
                                  // 1552B rows => 4-bank skew => conflict-free cols

constexpr int PAR[NJ] = {-1,0,0,0,1,2,3,4,5,6,7,8,9,9,9,12,13,14,16,17,18,19,20,21};

// ---------------------------------------------------------------------------
// Stage 1: one wave per graph; lane = hidden channel h in [0,32)
// ---------------------------------------------------------------------------
__global__ __launch_bounds__(256) void gat_pre_kernel(
    const float* __restrict__ src, const float* __restrict__ W_pre,
    const float* __restrict__ b_pre, const float* __restrict__ W_gat,
    const float* __restrict__ att_src, const float* __restrict__ att_dst,
    const float* __restrict__ b_gat, bf16_t* __restrict__ X)
{
  __shared__ float xsh[8][NJ][HIDD];
  const int lane = threadIdx.x & 31;
  const int wv   = threadIdx.x >> 5;
  const int g    = blockIdx.x * 8 + wv;           // graph id in [0, 32768)

  // per-lane (per-channel) weights
  const float Wp0 = W_pre[0*HIDD + lane];
  const float Wp1 = W_pre[1*HIDD + lane];
  const float Wp2 = W_pre[2*HIDD + lane];
  const float bp  = b_pre[lane];
  float Wg[HIDD];
  #pragma unroll
  for (int k = 0; k < HIDD; ++k) Wg[k] = W_gat[k*HIDD + lane];  // column `lane`
  const float ash = att_src[lane], adh = att_dst[lane], bg = b_gat[lane];

  const float* s = src + (size_t)g * (NJ*3);

  // pre: x = tanh(p @ W_pre + b)  -> LDS (wave-local, no __syncthreads needed)
  #pragma unroll
  for (int j = 0; j < NJ; ++j) {
    float p0 = s[j*3+0], p1 = s[j*3+1], p2 = s[j*3+2];
    xsh[wv][j][lane] = tanhf(fmaf(p0, Wp0, fmaf(p1, Wp1, fmaf(p2, Wp2, bp))));
  }

  // xp = x @ W_gat (LDS broadcast reads), attention scalars via wave reduction
  float xp[NJ], as_[NJ], ad_[NJ];
  #pragma unroll
  for (int j = 0; j < NJ; ++j) {
    const float4* row = reinterpret_cast<const float4*>(&xsh[wv][j][0]);
    float acc = 0.f;
    #pragma unroll
    for (int k4 = 0; k4 < HIDD/4; ++k4) {
      float4 xv = row[k4];
      acc = fmaf(xv.x, Wg[4*k4+0], acc);
      acc = fmaf(xv.y, Wg[4*k4+1], acc);
      acc = fmaf(xv.z, Wg[4*k4+2], acc);
      acc = fmaf(xv.w, Wg[4*k4+3], acc);
    }
    xp[j] = acc;
    float vs = acc * ash, vd = acc * adh;
    #pragma unroll
    for (int m = 16; m >= 1; m >>= 1) {       // wave32 all-reduce
      vs += __shfl_xor(vs, m, 32);
      vd += __shfl_xor(vd, m, 32);
    }
    as_[j] = vs; ad_[j] = vd;
  }

  // softmax over {self, parent} incoming edges; g = attn @ xp + b_gat
  bf16_t* xo = X + (size_t)g * (NJ*HIDD);
  #pragma unroll
  for (int j = 0; j < NJ; ++j) {
    const int p = PAR[j];                      // compile-time constant
    float gout;
    if (p < 0) {
      gout = xp[j];                            // single self edge -> attn = 1
    } else {
      float es = ad_[j] + as_[j]; es = es > 0.f ? es : 0.2f*es;   // LeakyReLU(0.2)
      float ep = ad_[j] + as_[p]; ep = ep > 0.f ? ep : 0.2f*ep;
      float mx = fmaxf(es, ep);
      float ws = __expf(es - mx), wp = __expf(ep - mx);
      gout = (ws*xp[j] + wp*xp[p]) / (ws + wp);
    }
    xo[j*HIDD + lane] = (bf16_t)(gout + bg);
  }
}

// ---------------------------------------------------------------------------
// Stage 2: f32 -> bf16 weights, bias sum
// ---------------------------------------------------------------------------
__global__ void prep_kernel(const float* __restrict__ Wih, const float* __restrict__ Whh,
                            const float* __restrict__ bih, const float* __restrict__ bhh,
                            bf16_t* __restrict__ Wih_b, bf16_t* __restrict__ Whh_b,
                            float* __restrict__ bsum)
{
  int i = blockIdx.x * blockDim.x + threadIdx.x;
  if (i < G4*HL) {
    Wih_b[i] = (bf16_t)Wih[i];
    Whh_b[i] = (bf16_t)Whh[i];
  }
  if (i < G4) bsum[i] = bih[i] + bhh[i];
}

// ---------------------------------------------------------------------------
// Stage 3: persistent LSTM
// ---------------------------------------------------------------------------
// 16-contiguous-bf16 fragment load (two b128s, 16B alignment everywhere)
__device__ inline v16bf load_frag16(const bf16_t* p) {
  union { v16bf v; v8bf h[2]; } u;
  u.h[0] = *(const v8bf*)(p);
  u.h[1] = *(const v8bf*)(p + 8);
  return u.v;
}

__device__ inline float sigm(float x) { return 1.0f / (1.0f + __expf(-x)); }

#define WMMA_BF16(A, B, C) \
  __builtin_amdgcn_wmma_f32_16x16x32_bf16(false, (A), false, (B), (short)0, (C), false, false)

__global__ __launch_bounds__(128, 1) void lstm_kernel(
    const bf16_t* __restrict__ X,      // [B=NB*LSEQ][HL] bf16 (== [n][t][HL])
    const bf16_t* __restrict__ Wih,    // [G4][HL] bf16 (row = out, col = in)
    const bf16_t* __restrict__ Whh,    // [G4][HL] bf16
    const float*  __restrict__ bsum,   // [G4] = b_ih + b_hh
    bf16_t* __restrict__ hbuf,         // [2][NB][HL] bf16, zeroed before launch
    float* __restrict__ out, float* __restrict__ hn, float* __restrict__ cn,
    unsigned* __restrict__ ctr)        // zeroed before launch
{
  extern __shared__ bf16_t lds[];                 // [2][64][LDP]
  bf16_t* wih_l = lds;
  bf16_t* whh_l = lds + 64*LDP;

  const int tid  = threadIdx.x;
  const int lane = tid & 31;
  const int wv   = tid >> 5;                      // 0..3
  const int j0   = blockIdx.x * 16;               // hidden-unit chunk base

  // Stage this WG's 64 W_ih rows + 64 W_hh rows into LDS (reused all 256 steps).
  // LDS row r = gate(r>>4)*16 + unit(r&15)  ->  global row (r>>4)*HL + j0 + (r&15)
  for (int idx = tid; idx < 2*64*(HL/8); idx += 128) {
    int mat = idx >= 64*(HL/8);
    int rem = idx - mat*64*(HL/8);
    int r   = rem / (HL/8);
    int k8  = (rem % (HL/8)) * 8;
    int grow = (r >> 4)*HL + j0 + (r & 15);
    const bf16_t* sp = (mat ? Whh : Wih) + (size_t)grow*HL + k8;
    bf16_t*       dp = (mat ? whh_l : wih_l) + r*LDP + k8;
    *(v8bf*)dp = *(const v8bf*)sp;
  }
  __syncthreads();

  // wave -> 32x16 block: rows [32*wv, 32*wv+32), hidden units [j0, j0+16)
  const int csel  = lane & 15;                    // fragment row/col select
  const int ksel  = (lane >> 4) << 4;             // K half select (0 / 16)
  const int rb0   = wv * 32;                      // M-tile 0 row base
  const int rb1   = rb0 + 16;                     // M-tile 1 row base
  const int arow0 = rb0 + csel;                   // A-fragment batch rows
  const int arow1 = rb1 + csel;
  const int ccol  = j0 + csel;                    // C-fragment hidden unit
  const int cr0   = rb0 + ((lane >> 4) << 3);     // C-fragment row bases
  const int cr1   = rb1 + ((lane >> 4) << 3);

  const float bi  = bsum[0*HL + ccol];
  const float bf_ = bsum[1*HL + ccol];
  const float bg  = bsum[2*HL + ccol];
  const float bo  = bsum[3*HL + ccol];

  v8f c0 = {}, c1 = {};                            // cell state, VGPR-resident

  for (int t = 0; t < LSEQ; ++t) {
    v8f a00, a01, a02, a03, a10, a11, a12, a13;    // [Mtile][gate] accumulators
    #pragma unroll
    for (int r = 0; r < 8; ++r) {
      a00[r] = bi;  a01[r] = bf_;  a02[r] = bg;  a03[r] = bo;
      a10[r] = bi;  a11[r] = bf_;  a12[r] = bg;  a13[r] = bo;
    }

    // ---- input part: gates += X_t @ W_ih^T -------------------------------
    const bf16_t* xr0 = X + ((size_t)arow0*LSEQ + t)*HL + ksel;
    const bf16_t* xr1 = X + ((size_t)arow1*LSEQ + t)*HL + ksel;
    if (t + 1 < LSEQ) {
      __builtin_prefetch(xr0 + HL, 0, 3);          // next timestep rows
      __builtin_prefetch(xr1 + HL, 0, 3);
    }
    #pragma unroll 2
    for (int kk = 0; kk < HL; kk += 32) {
      v16bf A0 = load_frag16(xr0 + kk);
      v16bf A1 = load_frag16(xr1 + kk);
      v16bf b0 = load_frag16(&wih_l[(0*16 + csel)*LDP + kk + ksel]);
      v16bf b1 = load_frag16(&wih_l[(1*16 + csel)*LDP + kk + ksel]);
      v16bf b2 = load_frag16(&wih_l[(2*16 + csel)*LDP + kk + ksel]);
      v16bf b3 = load_frag16(&wih_l[(3*16 + csel)*LDP + kk + ksel]);
      a00 = WMMA_BF16(A0, b0, a00);  a01 = WMMA_BF16(A0, b1, a01);
      a02 = WMMA_BF16(A0, b2, a02);  a03 = WMMA_BF16(A0, b3, a03);
      a10 = WMMA_BF16(A1, b0, a10);  a11 = WMMA_BF16(A1, b1, a11);
      a12 = WMMA_BF16(A1, b2, a12);  a13 = WMMA_BF16(A1, b3, a13);
    }

    // ---- recurrent part: gates += h_{t-1} @ W_hh^T (double-buffered h) ---
    const bf16_t* hr0 = hbuf + ((size_t)(t & 1)*NB + arow0)*HL + ksel;
    const bf16_t* hr1 = hbuf + ((size_t)(t & 1)*NB + arow1)*HL + ksel;
    #pragma unroll 2
    for (int kk = 0; kk < HL; kk += 32) {
      v16bf A0 = load_frag16(hr0 + kk);
      v16bf A1 = load_frag16(hr1 + kk);
      v16bf b0 = load_frag16(&whh_l[(0*16 + csel)*LDP + kk + ksel]);
      v16bf b1 = load_frag16(&whh_l[(1*16 + csel)*LDP + kk + ksel]);
      v16bf b2 = load_frag16(&whh_l[(2*16 + csel)*LDP + kk + ksel]);
      v16bf b3 = load_frag16(&whh_l[(3*16 + csel)*LDP + kk + ksel]);
      a00 = WMMA_BF16(A0, b0, a00);  a01 = WMMA_BF16(A0, b1, a01);
      a02 = WMMA_BF16(A0, b2, a02);  a03 = WMMA_BF16(A0, b3, a03);
      a10 = WMMA_BF16(A1, b0, a10);  a11 = WMMA_BF16(A1, b1, a11);
      a12 = WMMA_BF16(A1, b2, a12);  a13 = WMMA_BF16(A1, b3, a13);
    }

    // ---- gate nonlinearity + state update (wave-local) -------------------
    bf16_t* hw = hbuf + (size_t)((t + 1) & 1)*NB*HL;
    #pragma unroll
    for (int r = 0; r < 8; ++r) {
      // M-tile 0
      {
        float ig = sigm(a00[r]);
        float fg = sigm(a01[r]);
        float gg = tanhf(a02[r]);
        float og = sigm(a03[r]);
        float c  = fg * c0[r] + ig * gg;
        float h  = og * tanhf(c);
        c0[r] = c;
        int n = cr0 + r;
        out[((size_t)n*LSEQ + t)*HL + ccol] = h;
        hw[(size_t)n*HL + ccol] = (bf16_t)h;
        if (t == LSEQ - 1) { hn[(size_t)n*HL + ccol] = h; cn[(size_t)n*HL + ccol] = c; }
      }
      // M-tile 1
      {
        float ig = sigm(a10[r]);
        float fg = sigm(a11[r]);
        float gg = tanhf(a12[r]);
        float og = sigm(a13[r]);
        float c  = fg * c1[r] + ig * gg;
        float h  = og * tanhf(c);
        c1[r] = c;
        int n = cr1 + r;
        out[((size_t)n*LSEQ + t)*HL + ccol] = h;
        hw[(size_t)n*HL + ccol] = (bf16_t)h;
        if (t == LSEQ - 1) { hn[(size_t)n*HL + ccol] = h; cn[(size_t)n*HL + ccol] = c; }
      }
    }

    // ---- grid barrier: all 48 WGs must publish h before next step --------
    if (t != LSEQ - 1) {
      __threadfence();
      __syncthreads();
      if (tid == 0) {
        __hip_atomic_fetch_add(ctr, 1u, __ATOMIC_RELEASE, __HIP_MEMORY_SCOPE_AGENT);
        unsigned target = (unsigned)(t + 1) * CWG;      // monotonic, no reset races
        while (__hip_atomic_load(ctr, __ATOMIC_ACQUIRE, __HIP_MEMORY_SCOPE_AGENT) < target)
          __builtin_amdgcn_s_sleep(2);
      }
      __syncthreads();
    }
  }
}

// ---------------------------------------------------------------------------
// Host launcher (graph-capture safe: only kernel launches + memset nodes)
// ---------------------------------------------------------------------------
extern "C" void kernel_launch(void* const* d_in, const int* in_sizes, int n_in,
                              void* d_out, int out_size, void* d_ws, size_t ws_size,
                              hipStream_t stream)
{
  const float* src   = (const float*)d_in[0];
  const float* W_pre = (const float*)d_in[1];
  const float* b_pre = (const float*)d_in[2];
  const float* W_gat = (const float*)d_in[3];
  const float* att_s = (const float*)d_in[4];
  const float* att_d = (const float*)d_in[5];
  const float* b_gat = (const float*)d_in[6];
  const float* W_ih  = (const float*)d_in[7];
  const float* W_hh  = (const float*)d_in[8];
  const float* b_ih  = (const float*)d_in[9];
  const float* b_hh  = (const float*)d_in[10];

  // workspace layout (bytes, 256-aligned): ~58 MB total
  char* ws = (char*)d_ws;
  unsigned* ctr  = (unsigned*)(ws);                          //      256 B
  bf16_t*   X    = (bf16_t*)(ws + 256);                      // 50331648 B
  bf16_t*   hbuf = (bf16_t*)(ws + 256 + 50331648);           //   393216 B
  bf16_t*   WihB = (bf16_t*)(ws + 256 + 50331648 + 393216);
  bf16_t*   WhhB = (bf16_t*)((char*)WihB + (size_t)G4*HL*2);
  float*    bsum = (float*)((char*)WhhB + (size_t)G4*HL*2);

  float* out = (float*)d_out;
  float* hn  = out + (size_t)NB*LSEQ*HL;
  float* cn  = hn + (size_t)NB*HL;

  // deterministic per-call init (required across graph replays)
  hipMemsetAsync(ctr, 0, 256, stream);
  hipMemsetAsync(hbuf, 0, (size_t)2*NB*HL*sizeof(bf16_t), stream);

  gat_pre_kernel<<<NGRAPH/8, 256, 0, stream>>>(src, W_pre, b_pre, W_gat,
                                               att_s, att_d, b_gat, X);
  prep_kernel<<<(G4*HL + 255)/256, 256, 0, stream>>>(W_ih, W_hh, b_ih, b_hh,
                                                     WihB, WhhB, bsum);

  const int lds_bytes = 2 * 64 * LDP * (int)sizeof(bf16_t);  // 198656 B < 320 KB/WGP
  hipFuncSetAttribute((const void*)lstm_kernel,
                      hipFuncAttributeMaxDynamicSharedMemorySize, lds_bytes);
  lstm_kernel<<<CWG, 128, lds_bytes, stream>>>(X, WihB, WhhB, bsum, hbuf,
                                               out, hn, cn, ctr);
}
